// Ember_3384434229254
// MI455X (gfx1250) — matrix operand order repaired and checked by reading, not compiled, not run on hardware
//
#include <hip/hip_runtime.h>
#include <hip/hip_bf16.h>

// ---------------------------------------------------------------------------
// GPT-2-like forward (6 layers, LIF-gated attention) for MI455X / gfx1250.
// All dense GEMMs *and* attention contractions run on v_wmma_f32_16x16x32_f16
// (wave32 WMMA, fp32 accumulate). GEMMs are double-buffered in LDS with
// global_prefetch for the weight stream. Attention tiles honor causality:
// each 16-query tile only touches key chunks <= its own position.
// ---------------------------------------------------------------------------

typedef __attribute__((ext_vector_type(16))) _Float16 v16h;
typedef __attribute__((ext_vector_type(8)))  float    v8f;

static constexpr int kL = 6;
static constexpr int kC = 768;
static constexpr int kH = 12;
static constexpr int kT = 1024;
static constexpr int kB = 8;
static constexpr int kV = 50257;
static constexpr int kM = kB * kT;      // 8192 activation rows

union FragU { v16h v; _Float16 h[16]; unsigned int u[8]; };

// ---------------------------------------------------------------------------
// Embedding: x[b,t,:] = wte[idx[b,t],:] + wpe[t,:]
// ---------------------------------------------------------------------------
__global__ __launch_bounds__(256) void embed_kernel(const int* __restrict__ idx,
                                                    const float* __restrict__ wte,
                                                    const float* __restrict__ wpe,
                                                    float* __restrict__ x) {
  int bt = blockIdx.x;
  int t  = bt % kT;
  int tok = idx[bt];
  const float* wr = wte + (size_t)tok * kC;
  const float* pr = wpe + (size_t)t * kC;
  float* xr = x + (size_t)bt * kC;
  for (int c = threadIdx.x; c < kC; c += 256) xr[c] = wr[c] + pr[c];
}

// ---------------------------------------------------------------------------
// LayerNorm over last dim (C=768), one block per row.
// ---------------------------------------------------------------------------
__global__ __launch_bounds__(256) void ln_kernel(const float* __restrict__ x,
                                                 const float* __restrict__ w,
                                                 const float* __restrict__ b,
                                                 float* __restrict__ out) {
  __shared__ float r1[256];
  __shared__ float r2[256];
  int row = blockIdx.x;
  int tid = threadIdx.x;
  const float* xr = x + (size_t)row * kC;
  float s = 0.f, sq = 0.f;
  for (int c = tid; c < kC; c += 256) { float v = xr[c]; s += v; sq += v * v; }
  r1[tid] = s; r2[tid] = sq; __syncthreads();
  for (int st = 128; st > 0; st >>= 1) {
    if (tid < st) { r1[tid] += r1[tid + st]; r2[tid] += r2[tid + st]; }
    __syncthreads();
  }
  float mean = r1[0] * (1.f / kC);
  float var  = r2[0] * (1.f / kC) - mean * mean;
  float inv  = rsqrtf(var + 1e-5f);
  float* orow = out + (size_t)row * kC;
  for (int c = tid; c < kC; c += 256)
    orow[c] = (xr[c] - mean) * inv * w[c] + b[c];
}

// ---------------------------------------------------------------------------
// WMMA GEMM: out[M,N] = A[M,K] * W[N,K]^T + bias[N]   (+ residual / gelu)
// Block tile 128x64, 8 wave32 waves (4x2), each wave 32x32 (2x2 WMMA tiles).
// Double-buffered LDS (one barrier per K-step) + global_prefetch_b8 ahead.
// EPI: 0 = bias, 1 = bias + residual, 2 = bias + exact GELU.
// M % 128 == 0, N % 64 == 0, K % 32 == 0 (true for every use here).
// ---------------------------------------------------------------------------
static constexpr int kLDT = 36;   // padded LDS stride (halves)

template <int EPI>
__global__ __launch_bounds__(256) void gemm_kernel(const float* __restrict__ A,
                                                   const float* __restrict__ W,
                                                   const float* __restrict__ bias,
                                                   const float* __restrict__ res,
                                                   float* __restrict__ out,
                                                   int M, int N, int K) {
  constexpr int BM = 128, BN = 64, BK = 32;
  __shared__ _Float16 As[2][BM * kLDT];
  __shared__ _Float16 Bs[2][BN * kLDT];

  const int tid  = threadIdx.x;
  const int lane = tid & 31;
  const int wid  = tid >> 5;
  const int wm   = (wid & 3) * 32;
  const int wn   = (wid >> 2) * 32;
  const int blockM = blockIdx.y * BM;
  const int blockN = blockIdx.x * BN;
  const int g = lane >> 4;
  const int ln16 = lane & 15;

  v8f acc[2][2];
  #pragma unroll
  for (int i = 0; i < 2; ++i)
    #pragma unroll
    for (int j = 0; j < 2; ++j)
      acc[i][j] = (v8f){0.f,0.f,0.f,0.f,0.f,0.f,0.f,0.f};

  // A loader: 2 threads/row, 16 floats each. B loader: 4 threads/row, 8 floats.
  const int ar  = tid >> 1;
  const int ak0 = (tid & 1) * 16;
  const int br  = tid >> 2;
  const int bk0 = (tid & 3) * 8;
  const float* arow = A + (size_t)(blockM + ar) * K + ak0;
  const float* brow = W + (size_t)(blockN + br) * K + bk0;

  float4 aR[4], bR[2];
  auto stage = [&](int bufi) {
    _Float16* da = &As[bufi][ar * kLDT + ak0];
    #pragma unroll
    for (int j = 0; j < 4; ++j) {
      float4 f = aR[j];
      da[j * 4 + 0] = (_Float16)f.x; da[j * 4 + 1] = (_Float16)f.y;
      da[j * 4 + 2] = (_Float16)f.z; da[j * 4 + 3] = (_Float16)f.w;
    }
    _Float16* db = &Bs[bufi][br * kLDT + bk0];
    #pragma unroll
    for (int j = 0; j < 2; ++j) {
      float4 f = bR[j];
      db[j * 4 + 0] = (_Float16)f.x; db[j * 4 + 1] = (_Float16)f.y;
      db[j * 4 + 2] = (_Float16)f.z; db[j * 4 + 3] = (_Float16)f.w;
    }
  };

  // prologue: tile 0
  {
    const float4* sa = (const float4*)arow;
    const float4* sb = (const float4*)brow;
    #pragma unroll
    for (int j = 0; j < 4; ++j) aR[j] = sa[j];
    #pragma unroll
    for (int j = 0; j < 2; ++j) bR[j] = sb[j];
    stage(0);
  }
  __syncthreads();

  const int nsteps = K / BK;
  for (int s = 0; s < nsteps; ++s) {
    const int buf = s & 1;
    const bool more = (s + 1 < nsteps);
    if (more) {
      const float4* sa = (const float4*)(arow + (s + 1) * BK);
      const float4* sb = (const float4*)(brow + (s + 1) * BK);
      #pragma unroll
      for (int j = 0; j < 4; ++j) aR[j] = sa[j];
      #pragma unroll
      for (int j = 0; j < 2; ++j) bR[j] = sb[j];
      if (s + 2 < nsteps) {           // pull the step-after-next into cache
        __builtin_prefetch(arow + (s + 2) * BK, 0, 0);
        __builtin_prefetch(brow + (s + 2) * BK, 0, 0);
      }
    }

    // per-lane fragment gathers (ISA 7.12.2 layouts)
    FragU a[2], b[2];
    #pragma unroll
    for (int m16 = 0; m16 < 2; ++m16) {
      const _Float16* ap = &As[buf][(wm + m16 * 16 + ln16) * kLDT];
      #pragma unroll
      for (int j = 0; j < 4; ++j)
        a[m16].u[j] = *(const unsigned int*)(ap + g * 8 + 2 * j);
      #pragma unroll
      for (int j = 0; j < 4; ++j)
        a[m16].u[4 + j] = *(const unsigned int*)(ap + 16 + g * 8 + 2 * j);
    }
    #pragma unroll
    for (int n16 = 0; n16 < 2; ++n16) {
      const _Float16* bp = &Bs[buf][(wn + n16 * 16 + ln16) * kLDT + g * 8];
      #pragma unroll
      for (int j = 0; j < 8; ++j)
        b[n16].u[j] = *(const unsigned int*)(bp + (j >> 2) * 16 + 2 * (j & 3));
    }

    #pragma unroll
    for (int m16 = 0; m16 < 2; ++m16)
      #pragma unroll
      for (int n16 = 0; n16 < 2; ++n16)
        acc[m16][n16] = __builtin_amdgcn_wmma_f32_16x16x32_f16(
            false, a[m16].v, false, b[n16].v, (short)0, acc[m16][n16],
            false, false);

    if (more) stage(buf ^ 1);
    __syncthreads();
  }

  // epilogue: C/D layout M = r + 8*g, N = lane&15
  #pragma unroll
  for (int m16 = 0; m16 < 2; ++m16) {
    #pragma unroll
    for (int n16 = 0; n16 < 2; ++n16) {
      int col  = blockN + wn + n16 * 16 + ln16;
      int rowb = blockM + wm + m16 * 16 + g * 8;
      float bv = bias[col];
      #pragma unroll
      for (int r = 0; r < 8; ++r) {
        int row = rowb + r;
        float val = acc[m16][n16][r] + bv;
        if (EPI == 1) val += res[(size_t)row * N + col];
        if (EPI == 2) val = 0.5f * val * (1.f + erff(val * 0.70710678118654752f));
        out[(size_t)row * N + col] = val;
      }
    }
  }
}

// ---------------------------------------------------------------------------
// LIF-gated causal attention, fully on WMMA.
// One block (128 thr / 4 waves) per (b, h, 16-query tile).
//   Phase 1: S[16, nch*128] = Q(16x64) . K^T, WMMA, stored fp16 in LDS.
//   Phase 2: per-row masked softmax + LIF gate + renorm (shuffle reductions),
//            mod weights written back fp16 (P), 1/sum kept per row.
//   Phase 3: Y[16,64] = P . V via WMMA (V transposed into LDS per chunk).
// Causality: only key chunks covering k <= q0+15 are ever staged/computed.
// ---------------------------------------------------------------------------
__global__ __launch_bounds__(128) void attn_kernel(const float* __restrict__ qkv,
                                                   float* __restrict__ y,
                                                   const float* __restrict__ thr,
                                                   const float* __restrict__ leak,
                                                   const float* __restrict__ steep,
                                                   int layer) {
  constexpr int LDQ = 72, LDK = 72, LDV = 130, LDP = 1032;
  __shared__ _Float16 Qs[16 * LDQ];
  __shared__ _Float16 KV[128 * LDK];    // K chunk [key][d]; reused as V^T [d][key]
  __shared__ _Float16 P[16 * LDP];      // scores, then mod weights (fp16)
  __shared__ float invS[16];

  const int q0 = blockIdx.x * 16;
  const int b  = blockIdx.y / kH;
  const int hh = blockIdx.y % kH;
  const int tid  = threadIdx.x;
  const int lane = tid & 31;
  const int wid  = tid >> 5;            // 0..3
  const int g = lane >> 4;
  const int ln16 = lane & 15;
  const size_t rs = 3 * kC;
  const float* base = qkv + (size_t)b * kT * rs + hh * 64;

  // key chunks needed for causal coverage of queries q0..q0+15
  const int nch = (q0 + 16 + 127) / 128;
  const int kLim = nch * 128;

  // ---- stage Q tile (fp16) ----
  for (int i = tid; i < 16 * 64; i += 128) {
    int r = i >> 6, d = i & 63;
    Qs[r * LDQ + d] = (_Float16)base[(size_t)(q0 + r) * rs + d];
  }
  __syncthreads();

  // Q fragments (two 32-wide d windows), reused across all key chunks
  FragU aq[2];
  #pragma unroll
  for (int hdw = 0; hdw < 2; ++hdw) {
    const _Float16* ap = &Qs[ln16 * LDQ + hdw * 32];
    #pragma unroll
    for (int j = 0; j < 4; ++j)
      aq[hdw].u[j] = *(const unsigned int*)(ap + g * 8 + 2 * j);
    #pragma unroll
    for (int j = 0; j < 4; ++j)
      aq[hdw].u[4 + j] = *(const unsigned int*)(ap + 16 + g * 8 + 2 * j);
  }

  // ---- phase 1: scores ----
  for (int c = 0; c < nch; ++c) {
    for (int i = tid; i < 128 * 64; i += 128) {
      int key = i >> 6, d = i & 63;
      KV[key * LDK + d] = (_Float16)base[(size_t)(c * 128 + key) * rs + kC + d];
    }
    __syncthreads();

    v8f sacc[2];
    #pragma unroll
    for (int n16 = 0; n16 < 2; ++n16)
      sacc[n16] = (v8f){0.f,0.f,0.f,0.f,0.f,0.f,0.f,0.f};

    #pragma unroll
    for (int n16 = 0; n16 < 2; ++n16) {
      #pragma unroll
      for (int hdw = 0; hdw < 2; ++hdw) {
        FragU bk;
        const _Float16* bp =
            &KV[(wid * 32 + n16 * 16 + ln16) * LDK + hdw * 32 + g * 8];
        #pragma unroll
        for (int j = 0; j < 8; ++j)
          bk.u[j] = *(const unsigned int*)(bp + (j >> 2) * 16 + 2 * (j & 3));
        sacc[n16] = __builtin_amdgcn_wmma_f32_16x16x32_f16(
            false, aq[hdw].v, false, bk.v, (short)0, sacc[n16], false, false);
      }
    }
    #pragma unroll
    for (int n16 = 0; n16 < 2; ++n16) {
      int col = c * 128 + wid * 32 + n16 * 16 + ln16;
      #pragma unroll
      for (int r = 0; r < 8; ++r)
        P[(g * 8 + r) * LDP + col] = (_Float16)sacc[n16][r];
    }
    __syncthreads();
  }

  // ---- phase 2: masked softmax + LIF gate + renorm (per row) ----
  {
    float th  = fabsf(thr[layer * kH + hh]) * 0.1f;
    float lk  = 1.f / (1.f + __expf(-leak[layer * kH + hh]));
    float spv = steep[layer * kH + hh];
    float stp = (spv > 20.f) ? spv : log1pf(__expf(spv));
    for (int rr = wid * 4; rr < wid * 4 + 4; ++rr) {
      int qg = q0 + rr;
      _Float16* row = &P[rr * LDP];
      float mx = -1e30f;
      for (int k = lane; k < kLim; k += 32)
        if (k <= qg) mx = fmaxf(mx, (float)row[k] * 0.125f);
      #pragma unroll
      for (int off = 16; off > 0; off >>= 1)
        mx = fmaxf(mx, __shfl_xor(mx, off, 32));
      float z = 0.f;
      for (int k = lane; k < kLim; k += 32)
        if (k <= qg) z += __expf((float)row[k] * 0.125f - mx);
      #pragma unroll
      for (int off = 16; off > 0; off >>= 1) z += __shfl_xor(z, off, 32);
      float invZ = 1.f / z;
      float sm = 0.f;
      for (int k = lane; k < kLim; k += 32) {
        float m = 0.f;
        if (k <= qg) {
          float p = __expf((float)row[k] * 0.125f - mx) * invZ;
          float fire = 1.f / (1.f + __expf(-stp * (p - th)));
          m = p * (fire + lk * (1.f - fire));
        }
        row[k] = (_Float16)m;
        sm += m;
      }
      #pragma unroll
      for (int off = 16; off > 0; off >>= 1) sm += __shfl_xor(sm, off, 32);
      if (lane == 0) invS[rr] = 1.f / (sm + 1e-8f);
    }
  }
  __syncthreads();

  // ---- phase 3: Y = P . V ----
  v8f yacc = (v8f){0.f,0.f,0.f,0.f,0.f,0.f,0.f,0.f};
  for (int c = 0; c < nch; ++c) {
    for (int i = tid; i < 128 * 64; i += 128) {
      int key = i >> 6, d = i & 63;
      KV[d * LDV + key] =
          (_Float16)base[(size_t)(c * 128 + key) * rs + 2 * kC + d];
    }
    __syncthreads();
    #pragma unroll
    for (int s = 0; s < 4; ++s) {
      FragU pa, bv;
      const _Float16* pp = &P[ln16 * LDP + c * 128 + s * 32];
      #pragma unroll
      for (int j = 0; j < 4; ++j)
        pa.u[j] = *(const unsigned int*)(pp + g * 8 + 2 * j);
      #pragma unroll
      for (int j = 0; j < 4; ++j)
        pa.u[4 + j] = *(const unsigned int*)(pp + 16 + g * 8 + 2 * j);
      const _Float16* vp = &KV[(wid * 16 + ln16) * LDV + s * 32 + g * 8];
      #pragma unroll
      for (int j = 0; j < 8; ++j)
        bv.u[j] = *(const unsigned int*)(vp + (j >> 2) * 16 + 2 * (j & 3));
      yacc = __builtin_amdgcn_wmma_f32_16x16x32_f16(
          false, pa.v, false, bv.v, (short)0, yacc, false, false);
    }
    __syncthreads();
  }

  #pragma unroll
  for (int r = 0; r < 8; ++r) {
    int row = g * 8 + r;
    y[((size_t)(b * kT + q0 + row)) * kC + hh * 64 + wid * 16 + ln16] =
        yacc[r] * invS[row];
  }
}

// ---------------------------------------------------------------------------
// lm_head (tied): logits[b, v] = x_final[b, T-1, :] . wte[v, :]
// ---------------------------------------------------------------------------
__global__ __launch_bounds__(256) void logits_kernel(const float* __restrict__ xf,
                                                     const float* __restrict__ wte,
                                                     float* __restrict__ out) {
  __shared__ float xs[kB * kC];                 // 24 KB
  for (int i = threadIdx.x; i < kB * kC; i += 256) {
    int bb = i / kC, c = i % kC;
    xs[i] = xf[((size_t)(bb * kT + (kT - 1))) * kC + c];
  }
  __syncthreads();

  int wid  = threadIdx.x >> 5;
  int lane = threadIdx.x & 31;
  int v = blockIdx.x * 8 + wid;
  if (v >= kV) return;

  float acc[kB];
  #pragma unroll
  for (int bb = 0; bb < kB; ++bb) acc[bb] = 0.f;
  const float* wr = wte + (size_t)v * kC;
  for (int k = lane; k < kC; k += 32) {
    float wv = wr[k];
    #pragma unroll
    for (int bb = 0; bb < kB; ++bb) acc[bb] += wv * xs[bb * kC + k];
  }
  #pragma unroll
  for (int bb = 0; bb < kB; ++bb) {
    float a = acc[bb];
    for (int off = 16; off > 0; off >>= 1) a += __shfl_down(a, off, 32);
    if (lane == 0) out[(size_t)bb * kV + v] = a;
  }
}

// ---------------------------------------------------------------------------
// Host orchestration
// ---------------------------------------------------------------------------
extern "C" void kernel_launch(void* const* d_in, const int* in_sizes, int n_in,
                              void* d_out, int out_size, void* d_ws, size_t ws_size,
                              hipStream_t stream) {
  const int*   idx     = (const int*)  d_in[0];
  const float* wte     = (const float*)d_in[1];
  const float* wpe     = (const float*)d_in[2];
  const float* ln1_w   = (const float*)d_in[3];
  const float* ln1_b   = (const float*)d_in[4];
  const float* attn_w  = (const float*)d_in[5];
  const float* attn_b  = (const float*)d_in[6];
  const float* proj_w  = (const float*)d_in[7];
  const float* proj_b  = (const float*)d_in[8];
  const float* thr     = (const float*)d_in[9];
  const float* leak    = (const float*)d_in[10];
  const float* steep   = (const float*)d_in[11];
  const float* ln2_w   = (const float*)d_in[12];
  const float* ln2_b   = (const float*)d_in[13];
  const float* fc_w    = (const float*)d_in[14];
  const float* fc_b    = (const float*)d_in[15];
  const float* mproj_w = (const float*)d_in[16];
  const float* mproj_b = (const float*)d_in[17];
  const float* lnf_w   = (const float*)d_in[18];
  const float* lnf_b   = (const float*)d_in[19];
  float* out = (float*)d_out;

  float* ws  = (float*)d_ws;
  const size_t szAct = (size_t)kM * kC;
  float* x   = ws;
  float* h   = ws + szAct;
  float* y   = ws + 2 * szAct;
  float* big = ws + 3 * szAct;                  // [M, 4C] scratch (qkv / mlp)

  dim3 blk(256);

  embed_kernel<<<dim3(kM), blk, 0, stream>>>(idx, wte, wpe, x);

  for (int l = 0; l < kL; ++l) {
    ln_kernel<<<dim3(kM), blk, 0, stream>>>(x, ln1_w + l * kC, ln1_b + l * kC, h);
    gemm_kernel<0><<<dim3((3 * kC) / 64, kM / 128), blk, 0, stream>>>(
        h, attn_w + (size_t)l * 3 * kC * kC, attn_b + l * 3 * kC,
        nullptr, big, kM, 3 * kC, kC);
    attn_kernel<<<dim3(kT / 16, kB * kH), dim3(128), 0, stream>>>(
        big, y, thr, leak, steep, l);
    gemm_kernel<1><<<dim3(kC / 64, kM / 128), blk, 0, stream>>>(
        y, proj_w + (size_t)l * kC * kC, proj_b + l * kC, x, x, kM, kC, kC);
    ln_kernel<<<dim3(kM), blk, 0, stream>>>(x, ln2_w + l * kC, ln2_b + l * kC, h);
    gemm_kernel<2><<<dim3((4 * kC) / 64, kM / 128), blk, 0, stream>>>(
        h, fc_w + (size_t)l * 4 * kC * kC, fc_b + l * 4 * kC,
        nullptr, big, kM, 4 * kC, kC);
    gemm_kernel<1><<<dim3(kC / 64, kM / 128), blk, 0, stream>>>(
        big, mproj_w + (size_t)l * kC * 4 * kC, mproj_b + l * kC, x, x,
        kM, kC, 4 * kC);
  }

  ln_kernel<<<dim3(kM), blk, 0, stream>>>(x, lnf_w, lnf_b, h);
  logits_kernel<<<dim3((kV + 7) / 8), blk, 0, stream>>>(h, wte, out);
}
